// Encoder_63256278336085
// MI455X (gfx1250) — compile-verified, matrix operand order checked by reading, not run on hardware
//
#include <hip/hip_runtime.h>
#include <hip/hip_bf16.h>
#include <math.h>

typedef __bf16 bf16;
typedef __attribute__((ext_vector_type(16))) __bf16 v16bf;
typedef __attribute__((ext_vector_type(8)))  __bf16 v8bf;
typedef __attribute__((ext_vector_type(8)))  float  v8f;

#define B_   8
#define S_   512
#define H_   16
#define P_   64
#define I_   1024
#define U_   4096
#define M_   4096   // B*S

// ---------------------------------------------------------------------------
// WMMA + async helpers
// ---------------------------------------------------------------------------
__device__ __forceinline__ v8f wmma_bf16(v16bf a, v16bf b, v8f c) {
  return __builtin_amdgcn_wmma_f32_16x16x32_bf16(
      /*neg_a=*/false, a, /*neg_b=*/false, b,
      /*c_mod=*/(short)0, c, /*reuse_a=*/false, /*reuse_b=*/false);
}

// Build a 16x32 bf16 A/B fragment for one lane. `p` points at element
// [row][8*half] of a row-major K=32 slab (LDS or global).
// elems 0..7 -> k = 8*half + 0..7 ; elems 8..15 -> k = 16 + 8*half + 0..7
__device__ __forceinline__ v16bf make_frag(const bf16* p) {
  v8bf lo = *(const v8bf*)(p);
  v8bf hi = *(const v8bf*)(p + 16);
  v16bf r;
#pragma unroll
  for (int i = 0; i < 8; ++i) { r[i] = lo[i]; r[i + 8] = hi[i]; }
  return r;
}

// Per-lane async 16B copy global -> LDS, tracked by ASYNCcnt.
__device__ __forceinline__ void async_ld_b128(void* lds_dst, const void* gsrc) {
  unsigned loff = (unsigned)(uintptr_t)lds_dst;  // low 32 bits = LDS offset
  asm volatile("global_load_async_to_lds_b128 %0, %1, off"
               :: "v"(loff), "v"(gsrc) : "memory");
}
__device__ __forceinline__ void wait_async0() {
  asm volatile("s_wait_asynccnt 0" ::: "memory");
}

// ---------------------------------------------------------------------------
// Conversion / layout kernels
// ---------------------------------------------------------------------------
__global__ __launch_bounds__(256) void cvt_f32_bf16(const float* __restrict__ in,
                                                    bf16* __restrict__ out, int n) {
  int idx = blockIdx.x * 256 + threadIdx.x;
  if (idx < n) out[idx] = (bf16)in[idx];
}

// in: [batch][K][N] f32  ->  out: [batch][N][K] bf16   (idx: k fastest)
__global__ __launch_bounds__(256) void wtrans(const float* __restrict__ in,
                                              bf16* __restrict__ out,
                                              int K, int N, int n) {
  long long idx = (long long)blockIdx.x * 256 + threadIdx.x;
  if (idx >= n) return;
  int k = (int)(idx % K);
  long long t = idx / K;
  int nn = (int)(t % N);
  long long bb = t / N;
  out[idx] = (bf16)in[(bb * K + k) * (long long)N + nn];
}

// vt[b][h][p][s] = qkv[(b*S+s)*3072 + 2048 + h*64 + p]
__global__ __launch_bounds__(256) void vtrans(const bf16* __restrict__ qkv,
                                              bf16* __restrict__ vt, int n) {
  int idx = blockIdx.x * 256 + threadIdx.x;
  if (idx >= n) return;
  int s = idx & (S_ - 1);
  int p = (idx >> 9) & (P_ - 1);
  int h = (idx >> 15) & (H_ - 1);
  int b = idx >> 19;
  vt[idx] = qkv[(size_t)(b * S_ + s) * 3072 + 2048 + h * 64 + p];
}

// ---------------------------------------------------------------------------
// Tiled bf16 GEMM with WMMA, double-buffered async LDS staging.
// C[M][N] = A[M][K] * Bt[N][K]^T + bias
// WG tile 128x128, 8 waves (4 along M x 2 along N), wave tile 32x64.
// EPI: 0 = bias -> bf16 ; 1 = bias+ReLU -> bf16 ; 2 = bias -> f32
// ---------------------------------------------------------------------------
template <int EPI>
__global__ __launch_bounds__(256) void gemm_bf16_kernel(
    const bf16* __restrict__ A, const bf16* __restrict__ Bt,
    const float* __restrict__ bias, void* __restrict__ Cout,
    int M, int N, int K) {
  __shared__ __align__(16) bf16 lA[2][128][40];  // +16B pad vs bank conflicts
  __shared__ __align__(16) bf16 lB[2][128][40];

  const int tid  = threadIdx.x;
  const int lane = tid & 31;
  const int wave = tid >> 5;
  const int half = lane >> 4;
  const int l16  = lane & 15;
  const int wm   = wave & 3;   // 0..3 along M (32 rows each)
  const int wn   = wave >> 2;  // 0..1 along N (64 cols each)
  const int m0   = blockIdx.y * 128;
  const int n0   = blockIdx.x * 128;

  v8f acc[2][4];
#pragma unroll
  for (int i = 0; i < 2; ++i)
#pragma unroll
    for (int j = 0; j < 4; ++j)
#pragma unroll
      for (int r = 0; r < 8; ++r) acc[i][j][r] = 0.0f;

  const int arow = tid >> 1, aseg = tid & 1;  // 2 x 16B chunks per thread

  const bf16* gA = A + (size_t)(m0 + arow) * K;
  const bf16* gB = Bt + (size_t)(n0 + arow) * K;

  // prologue: stage tile 0
#pragma unroll
  for (int j = 0; j < 2; ++j) {
    int seg = aseg * 2 + j;
    async_ld_b128(&lA[0][arow][seg * 8], gA + seg * 8);
    async_ld_b128(&lB[0][arow][seg * 8], gB + seg * 8);
  }
  wait_async0();
  __syncthreads();

  int cur = 0;
  for (int k0 = 0; k0 < K; k0 += 32) {
    int nxt = cur ^ 1;
    if (k0 + 32 < K) {  // issue next tile while computing this one
#pragma unroll
      for (int j = 0; j < 2; ++j) {
        int seg = aseg * 2 + j;
        async_ld_b128(&lA[nxt][arow][seg * 8], gA + k0 + 32 + seg * 8);
        async_ld_b128(&lB[nxt][arow][seg * 8], gB + k0 + 32 + seg * 8);
      }
      if (k0 + 64 < K) {  // warm L2 one tile further ahead
        __builtin_prefetch(gA + k0 + 64, 0, 1);
        __builtin_prefetch(gB + k0 + 64, 0, 1);
      }
    }

    v16bf af[2], bfg[4];
#pragma unroll
    for (int t = 0; t < 2; ++t)
      af[t] = make_frag(&lA[cur][wm * 32 + t * 16 + l16][half * 8]);
#pragma unroll
    for (int t = 0; t < 4; ++t)
      bfg[t] = make_frag(&lB[cur][wn * 64 + t * 16 + l16][half * 8]);
#pragma unroll
    for (int ti = 0; ti < 2; ++ti)
#pragma unroll
      for (int tj = 0; tj < 4; ++tj)
        acc[ti][tj] = wmma_bf16(af[ti], bfg[tj], acc[ti][tj]);

    wait_async0();
    __syncthreads();
    cur = nxt;
  }

  // Epilogue.  D layout: lane holds col l16; VGPR r holds row r + 8*half.
#pragma unroll
  for (int ti = 0; ti < 2; ++ti) {
#pragma unroll
    for (int tj = 0; tj < 4; ++tj) {
      int col = n0 + wn * 64 + tj * 16 + l16;
      float bv = bias[col];
#pragma unroll
      for (int r = 0; r < 8; ++r) {
        int row = m0 + wm * 32 + ti * 16 + r + half * 8;
        float v = acc[ti][tj][r] + bv;
        if (EPI == 1) v = fmaxf(v, 0.0f);
        if (EPI == 2) {
          ((float*)Cout)[(size_t)row * N + col] = v;
        } else {
          ((bf16*)Cout)[(size_t)row * N + col] = (bf16)v;
        }
      }
    }
  }
}

// ---------------------------------------------------------------------------
// Fused attention (flash-style, online softmax), double-buffered cooperative
// K/V staging.  Grid: B*H*(S/128) blocks, 256 threads; wave = 16 query rows.
// qkv: [M][3072] bf16 (K cols 0..1023, Q at 1024, V at 2048)
// vt : [B][H][P][S] bf16 ;  ctx: [M][I] bf16
// ---------------------------------------------------------------------------
__global__ __launch_bounds__(256) void attn_kernel(
    const bf16* __restrict__ qkv, const bf16* __restrict__ vt,
    bf16* __restrict__ ctxb) {
  __shared__ __align__(16) bf16  lK[2][32][72];    // 32 keys x 64 (P)  (+pad)
  __shared__ __align__(16) bf16  lV[2][64][40];    // 64 (P) x 32 keys  (+pad)
  __shared__ __align__(16) float sT[8][16][36];    // per-wave raw scores
  __shared__ __align__(16) float sStat[8][16];     // per-wave row broadcast
  __shared__ __align__(16) bf16  lP[8][16][40];    // per-wave P tile

  const int tid  = threadIdx.x;
  const int lane = tid & 31;
  const int wave = tid >> 5;
  const int half = lane >> 4;
  const int l16  = lane & 15;

  const int blk = blockIdx.x;
  const int jb  = blk & 3;
  const int h   = (blk >> 2) & (H_ - 1);
  const int b   = blk >> 6;
  const int j0  = jb * 128 + wave * 16;
  const int ldq = 3072;

  // Q fragments for this wave's 16 rows (two K=32 steps over P=64)
  const bf16* qrow = qkv + (size_t)(b * S_ + j0 + l16) * ldq + 1024 + h * 64;
  v16bf qf0 = make_frag(qrow + half * 8);
  v16bf qf1 = make_frag(qrow + 32 + half * 8);

  float mrow = -1e30f, lrow = 0.0f;  // running stats for row l16 (dup halves)
  v8f cacc[4];
#pragma unroll
  for (int t = 0; t < 4; ++t)
#pragma unroll
    for (int r = 0; r < 8; ++r) cacc[t][r] = 0.0f;

  const int krow_l = tid >> 3, kseg = tid & 7;  // K tile: 32 rows x 8 chunks
  const int vrow_l = tid >> 2, vseg = tid & 3;  // V tile: 64 rows x 4 chunks

  const bf16* kbase = qkv + (size_t)(b * S_) * ldq + h * 64;  // K at col 0
  const bf16* vbase = vt + (size_t)((b * H_ + h) * P_) * S_;

  // prologue: stage key-block 0
  async_ld_b128(&lK[0][krow_l][kseg * 8],
                kbase + (size_t)krow_l * ldq + kseg * 8);
  async_ld_b128(&lV[0][vrow_l][vseg * 8],
                vbase + (size_t)vrow_l * S_ + vseg * 8);
  wait_async0();
  __syncthreads();

  int cur = 0;
  for (int kb = 0; kb < S_ / 32; ++kb) {
    int nxt = cur ^ 1;
    if (kb + 1 < S_ / 32) {  // issue next block while computing this one
      async_ld_b128(&lK[nxt][krow_l][kseg * 8],
                    kbase + (size_t)((kb + 1) * 32 + krow_l) * ldq + kseg * 8);
      async_ld_b128(&lV[nxt][vrow_l][vseg * 8],
                    vbase + (size_t)vrow_l * S_ + (kb + 1) * 32 + vseg * 8);
    }

    // ---- scores: two 16x16 tiles covering 32 key positions ----
    v8f sc[2];
#pragma unroll
    for (int cb = 0; cb < 2; ++cb) {
#pragma unroll
      for (int r = 0; r < 8; ++r) sc[cb][r] = 0.0f;
      v16bf kf0 = make_frag(&lK[cur][cb * 16 + l16][half * 8]);
      v16bf kf1 = make_frag(&lK[cur][cb * 16 + l16][32 + half * 8]);
      sc[cb] = wmma_bf16(qf0, kf0, sc[cb]);
      sc[cb] = wmma_bf16(qf1, kf1, sc[cb]);
    }

    // ---- transpose scores through LDS: lane -> one row ----
#pragma unroll
    for (int cb = 0; cb < 2; ++cb)
#pragma unroll
      for (int r = 0; r < 8; ++r)
        sT[wave][r + half * 8][cb * 16 + l16] = sc[cb][r] * 0.125f;

    float x[16];
    {
      const float4* t4 = (const float4*)&sT[wave][l16][half * 16];
#pragma unroll
      for (int i = 0; i < 4; ++i) {
        float4 v = t4[i];
        x[4 * i + 0] = v.x; x[4 * i + 1] = v.y;
        x[4 * i + 2] = v.z; x[4 * i + 3] = v.w;
      }
    }
    float rm = x[0];
#pragma unroll
    for (int i = 1; i < 16; ++i) rm = fmaxf(rm, x[i]);
    rm = fmaxf(rm, __shfl_xor(rm, 16, 32));   // combine the two half-rows
    float mn = fmaxf(mrow, rm);
    float al = __expf(mrow - mn);
    float rs = 0.0f;
#pragma unroll
    for (int i = 0; i < 16; ++i) { x[i] = __expf(x[i] - mn); rs += x[i]; }
    rs += __shfl_xor(rs, 16, 32);
    lrow = lrow * al + rs;
    mrow = mn;

    // broadcast per-row alpha for D-layout rescale
    if (half == 0) sStat[wave][l16] = al;

    // P tile back to LDS in row-major (A-frag friendly)
    {
      v8bf p0, p1;
#pragma unroll
      for (int i = 0; i < 8; ++i) { p0[i] = (bf16)x[i]; p1[i] = (bf16)x[i + 8]; }
      *(v8bf*)&lP[wave][l16][half * 16]     = p0;
      *(v8bf*)&lP[wave][l16][half * 16 + 8] = p1;
    }

    // rescale accumulators with per-row alpha (rows r + 8*half)
    {
      const float4* a4 = (const float4*)&sStat[wave][half * 8];
      float4 a0 = a4[0], a1 = a4[1];
      float av[8] = {a0.x, a0.y, a0.z, a0.w, a1.x, a1.y, a1.z, a1.w};
#pragma unroll
      for (int t = 0; t < 4; ++t)
#pragma unroll
        for (int r = 0; r < 8; ++r) cacc[t][r] *= av[r];
    }

    // ---- ctx += P (16x32) * V (32x64) ----
    v16bf pf = make_frag(&lP[wave][l16][half * 8]);
#pragma unroll
    for (int t = 0; t < 4; ++t) {
      v16bf vf = make_frag(&lV[cur][t * 16 + l16][half * 8]);
      cacc[t] = wmma_bf16(pf, vf, cacc[t]);
    }

    wait_async0();
    __syncthreads();
    cur = nxt;
  }

  // ---- finalize: divide by row sums, store ctx in [M][I] layout ----
  if (half == 0) sStat[wave][l16] = lrow;
  {
    const float4* a4 = (const float4*)&sStat[wave][half * 8];
    float4 a0 = a4[0], a1 = a4[1];
    float lv[8] = {a0.x, a0.y, a0.z, a0.w, a1.x, a1.y, a1.z, a1.w};
#pragma unroll
    for (int t = 0; t < 4; ++t) {
#pragma unroll
      for (int r = 0; r < 8; ++r) {
        float v = cacc[t][r] / lv[r];
        int row = b * S_ + j0 + r + half * 8;
        ctxb[(size_t)row * I_ + h * 64 + t * 16 + l16] = (bf16)v;
      }
    }
  }
}

// ---------------------------------------------------------------------------
// Residual + LayerNorm.  One block per row of 1024.
// ---------------------------------------------------------------------------
__global__ __launch_bounds__(256) void ln_kernel(
    const float* __restrict__ a, const float* __restrict__ bres,
    const float* __restrict__ g, const float* __restrict__ bb,
    float* __restrict__ outf, bf16* __restrict__ outb) {
  __shared__ float red[256];
  const int row = blockIdx.x;
  const int tid = threadIdx.x;
  const float* pa = a + (size_t)row * I_;
  const float* pb = bres + (size_t)row * I_;

  float x[4];
  float s = 0.0f;
#pragma unroll
  for (int i = 0; i < 4; ++i) {
    x[i] = pa[tid + i * 256] + pb[tid + i * 256];
    s += x[i];
  }
  red[tid] = s;
  __syncthreads();
  for (int o = 128; o > 0; o >>= 1) {
    if (tid < o) red[tid] += red[tid + o];
    __syncthreads();
  }
  float mu = red[0] * (1.0f / I_);
  __syncthreads();

  float v = 0.0f;
#pragma unroll
  for (int i = 0; i < 4; ++i) {
    float d = x[i] - mu;
    v += d * d;
  }
  red[tid] = v;
  __syncthreads();
  for (int o = 128; o > 0; o >>= 1) {
    if (tid < o) red[tid] += red[tid + o];
    __syncthreads();
  }
  float rstd = rsqrtf(red[0] * (1.0f / I_) + 1e-5f);

#pragma unroll
  for (int i = 0; i < 4; ++i) {
    int c = tid + i * 256;
    float y = (x[i] - mu) * rstd * g[c] + bb[c];
    outf[(size_t)row * I_ + c] = y;
    if (outb) outb[(size_t)row * I_ + c] = (bf16)y;
  }
}

// ---------------------------------------------------------------------------
// Host-side orchestration
// ---------------------------------------------------------------------------
extern "C" void kernel_launch(void* const* d_in, const int* in_sizes, int n_in,
                              void* d_out, int out_size, void* d_ws, size_t ws_size,
                              hipStream_t stream) {
  (void)in_sizes; (void)n_in; (void)out_size; (void)ws_size;

  const float* X     = (const float*)d_in[0];   // [B,S,I]
  const float* WKQV  = (const float*)d_in[1];   // [3,I,I]
  const float* BKQV  = (const float*)d_in[2];   // [3,I]
  const float* WO    = (const float*)d_in[3];   // [I,I]
  const float* BO    = (const float*)d_in[4];   // [I]
  const float* ln1g  = (const float*)d_in[5];
  const float* ln1b  = (const float*)d_in[6];
  const float* W1    = (const float*)d_in[7];   // [I,U]
  const float* B1    = (const float*)d_in[8];   // [U]
  const float* W2    = (const float*)d_in[9];   // [U,I]
  const float* B2    = (const float*)d_in[10];  // [I]
  const float* ln2g  = (const float*)d_in[11];
  const float* ln2b  = (const float*)d_in[12];
  float* OUT = (float*)d_out;

  char* ws = (char*)d_ws;
  size_t off = 0;
  auto take = [&](size_t bytes) {
    size_t o = off;
    off += (bytes + 255) & ~(size_t)255;
    return o;
  };
  bf16*  XB    = (bf16*) (ws + take((size_t)M_ * I_ * 2));
  bf16*  WKQVT = (bf16*) (ws + take((size_t)3 * I_ * I_ * 2));   // [3*I][I]
  bf16*  WOT   = (bf16*) (ws + take((size_t)I_ * I_ * 2));       // [I][I]
  bf16*  W1T   = (bf16*) (ws + take((size_t)U_ * I_ * 2));       // [U][I]
  bf16*  W2T   = (bf16*) (ws + take((size_t)I_ * U_ * 2));       // [I][U]
  bf16*  QKV   = (bf16*) (ws + take((size_t)M_ * 3072 * 2));     // [M][3I]
  bf16*  VT    = (bf16*) (ws + take((size_t)B_ * H_ * P_ * S_ * 2));
  bf16*  CTX   = (bf16*) (ws + take((size_t)M_ * I_ * 2));
  float* AO    = (float*)(ws + take((size_t)M_ * I_ * 4));
  float* H1F   = (float*)(ws + take((size_t)M_ * I_ * 4));
  bf16*  H1B   = (bf16*) (ws + take((size_t)M_ * I_ * 2));
  bf16*  FF1   = (bf16*) (ws + take((size_t)M_ * U_ * 2));
  float* FFO   = AO;  // attn_out dead after LN1

  { int n = M_ * I_;     cvt_f32_bf16<<<n / 256, 256, 0, stream>>>(X, XB, n); }
  { int n = 3 * I_ * I_; wtrans<<<n / 256, 256, 0, stream>>>(WKQV, WKQVT, I_, I_, n); }
  { int n = I_ * I_;     wtrans<<<n / 256, 256, 0, stream>>>(WO, WOT, I_, I_, n); }
  { int n = I_ * U_;     wtrans<<<n / 256, 256, 0, stream>>>(W1, W1T, I_, U_, n); }
  { int n = U_ * I_;     wtrans<<<n / 256, 256, 0, stream>>>(W2, W2T, U_, I_, n); }

  // QKV projection: [M][3072] = XB * WKQVT^T + BKQV
  gemm_bf16_kernel<0><<<dim3(3072 / 128, M_ / 128), 256, 0, stream>>>(
      XB, WKQVT, BKQV, (void*)QKV, M_, 3072, I_);
  { int n = B_ * H_ * P_ * S_; vtrans<<<n / 256, 256, 0, stream>>>(QKV, VT, n); }

  attn_kernel<<<B_ * H_ * (S_ / 128), 256, 0, stream>>>(QKV, VT, CTX);

  gemm_bf16_kernel<2><<<dim3(I_ / 128, M_ / 128), 256, 0, stream>>>(
      CTX, WOT, BO, (void*)AO, M_, I_, I_);
  ln_kernel<<<M_, 256, 0, stream>>>(X, AO, ln1g, ln1b, H1F, H1B);

  gemm_bf16_kernel<1><<<dim3(U_ / 128, M_ / 128), 256, 0, stream>>>(
      H1B, W1T, B1, (void*)FF1, M_, U_, I_);
  gemm_bf16_kernel<2><<<dim3(I_ / 128, M_ / 128), 256, 0, stream>>>(
      FF1, W2T, B2, (void*)FFO, M_, I_, U_);
  ln_kernel<<<M_, 256, 0, stream>>>(H1F, FFO, ln2g, ln2b, OUT, nullptr);
}
